// BitBalanceHardMiningLoss_43087111914075
// MI455X (gfx1250) — compile-verified, matrix operand order checked by reading, not run on hardware
//
#include <hip/hip_runtime.h>
#include <stdint.h>

#define NROWS 32
#define LROW  (768*768)           // 589824 elements per row
#define CHUNK 2048                // elements per workgroup in streaming kernels
#define NCHUNK (LROW / CHUNK)     // 288

typedef _Float16 v16h __attribute__((ext_vector_type(16)));
typedef float    v8f  __attribute__((ext_vector_type(8)));

struct Sel { unsigned pval, pmask, need, k; };

// ---------------- K0: zero the small scratch regions ----------------
__global__ void k_zero(unsigned* __restrict__ pos, unsigned* __restrict__ hist,
                       unsigned* __restrict__ cnt, float* __restrict__ part) {
  const int t = threadIdx.x;
  for (int i = t; i < NROWS;        i += 256) pos[i]  = 0u;
  for (int i = t; i < NROWS * 2048; i += 256) hist[i] = 0u;
  for (int i = t; i < NROWS * 2;    i += 256) cnt[i]  = 0u;
  for (int i = t; i < 128;          i += 256) part[i] = 0.f;
}

// ---------------- K1: cross entropy + positive counts ----------------
// ce = softplus(l_other - l_target); sign bit of stored value encodes target==1.
__global__ void k_ce(const float* __restrict__ logits, const long long* __restrict__ tgt,
                     float* __restrict__ ce, unsigned* __restrict__ pos_count) {
  const int row = blockIdx.y;
  const int e0  = (blockIdx.x * 256 + threadIdx.x) * 8;
  const size_t lbase = (size_t)row * 2 * LROW + e0;
  const float4* l0p = (const float4*)(logits + lbase);
  const float4* l1p = (const float4*)(logits + lbase + LROW);
  const long long* tp = tgt + (size_t)row * LROW + e0;

  float4 a0 = l0p[0], a1 = l0p[1];
  float4 b0 = l1p[0], b1 = l1p[1];
  const float A[8] = {a0.x, a0.y, a0.z, a0.w, a1.x, a1.y, a1.z, a1.w};
  const float B[8] = {b0.x, b0.y, b0.z, b0.w, b1.x, b1.y, b1.z, b1.w};

  unsigned pc = 0;
  float o[8];
#pragma unroll
  for (int j = 0; j < 8; ++j) {
    const bool pos = (tp[j] != 0);
    pc += pos ? 1u : 0u;
    const float z = pos ? (A[j] - B[j]) : (B[j] - A[j]);      // l_other - l_target
    const float cev = fmaxf(z, 0.f) + log1pf(__expf(-fabsf(z)));  // stable softplus
    unsigned bits = __float_as_uint(cev) & 0x7FFFFFFFu;
    if (pos) bits |= 0x80000000u;                              // stash target in sign bit
    o[j] = __uint_as_float(bits);
  }
  float4* op = (float4*)(ce + (size_t)row * LROW + e0);
  op[0] = make_float4(o[0], o[1], o[2], o[3]);
  op[1] = make_float4(o[4], o[5], o[6], o[7]);

#pragma unroll
  for (int off = 16; off >= 1; off >>= 1) pc += __shfl_xor(pc, off);
  __shared__ unsigned red[8];
  const int wave = threadIdx.x >> 5, lane = threadIdx.x & 31;
  if (lane == 0) red[wave] = pc;
  __syncthreads();
  if (threadIdx.x == 0) {
    unsigned s = 0;
    for (int i = 0; i < 8; ++i) s += red[i];
    atomicAdd(pos_count + row, s);
  }
}

// ---------------- K2: WMMA rowsum for rows 0 and 1 ----------------
// D = A x Ones + C : every column of D holds the row-sums of A, so sum(D) = 16*sum(A).
// f32 accuracy kept via hi/lo f16 split (two WMMAs per 512-element tile).
__global__ void k_rowsum_wmma(const float* __restrict__ ce, float* __restrict__ part) {
  const int row  = blockIdx.y;               // 0..1
  const int wave = threadIdx.x >> 5;
  const int lane = threadIdx.x & 31;
  const int gw   = blockIdx.x * 8 + wave;    // 0..383 global wave id

  v8f acc;
#pragma unroll
  for (int i = 0; i < 8; ++i) acc[i] = 0.f;
  v16h ones;
#pragma unroll
  for (int i = 0; i < 16; ++i) ones[i] = (_Float16)1.0f;

  const float* rp = ce + (size_t)row * LROW;
#pragma unroll
  for (int c = 0; c < 3; ++c) {              // 384 waves * 3 chunks * 512 = LROW
    const int chunk = gw * 3 + c;
    const float4* p = (const float4*)(rp + chunk * 512 + lane * 16);
    float4 v0 = p[0], v1 = p[1], v2 = p[2], v3 = p[3];
    const float f[16] = {v0.x, v0.y, v0.z, v0.w, v1.x, v1.y, v1.z, v1.w,
                         v2.x, v2.y, v2.z, v2.w, v3.x, v3.y, v3.z, v3.w};
    v16h hi, lo;
#pragma unroll
    for (int j = 0; j < 16; ++j) {
      const float x = fabsf(f[j]);           // strip target bit
      const _Float16 h = (_Float16)x;
      hi[j] = h;
      lo[j] = (_Float16)(x - (float)h);
    }
    acc = __builtin_amdgcn_wmma_f32_16x16x32_f16(false, hi, false, ones, (short)0, acc, false, false);
    acc = __builtin_amdgcn_wmma_f32_16x16x32_f16(false, lo, false, ones, (short)0, acc, false, false);
  }

  float s = 0.f;
#pragma unroll
  for (int i = 0; i < 8; ++i) s += acc[i];
#pragma unroll
  for (int off = 16; off >= 1; off >>= 1) s += __shfl_xor(s, off);
  __shared__ float red[8];
  if (lane == 0) red[wave] = s;
  __syncthreads();
  if (threadIdx.x == 0) {
    float t = 0.f;
    for (int i = 0; i < 8; ++i) t += red[i];
    part[row * 64 + blockIdx.x] = t;         // later divided by 16
  }
}

// ---------------- K3: init per-row selection state ----------------
__global__ void k_sel_init(const unsigned* __restrict__ pos_count, Sel* __restrict__ state) {
  const int r = threadIdx.x;
  if (r < NROWS) {
    const unsigned p = pos_count[r];
    const unsigned n = (unsigned)LROW - p;
    const unsigned k = p < n ? p : n;
    state[r] = Sel{0u, 0u, k, k};
  }
}

// ---------------- K4: radix-select histogram pass ----------------
// Tiles of ce are staged into LDS via the gfx1250 async-to-LDS path.
__global__ void k_hist(const float* __restrict__ ce, const Sel* __restrict__ state,
                       unsigned* __restrict__ ghist, int shift, unsigned bmask) {
  const int row = blockIdx.y;
  __shared__ float tile[CHUNK];
  __shared__ unsigned hs[2048];
  for (int b = threadIdx.x; b < 2048; b += 256) hs[b] = 0u;

  const Sel s = state[row];
  const unsigned goff = (unsigned)((blockIdx.x * CHUNK + threadIdx.x * 8) * 4);
  const unsigned lds0 = (unsigned)(uintptr_t)(&tile[0]) + (unsigned)threadIdx.x * 32u;
  const unsigned long long gbase =
      (unsigned long long)(uintptr_t)(ce + (size_t)row * LROW);
  // Async DMA 32B/thread (8KB/WG) into LDS; tracked by ASYNCcnt.
  asm volatile(
      "global_load_async_to_lds_b128 %0, %1, %2\n\t"
      "global_load_async_to_lds_b128 %0, %1, %2 offset:16\n\t"
      "s_wait_asynccnt 0"
      :: "v"(lds0), "v"(goff), "s"(gbase) : "memory");
  __syncthreads();

  const unsigned pm = s.pmask, pv = s.pval;
#pragma unroll
  for (int j = 0; j < 8; ++j) {
    const unsigned bits = __float_as_uint(tile[threadIdx.x * 8 + j]) & 0x7FFFFFFFu;
    if ((bits & pm) == pv) atomicAdd(&hs[(bits >> shift) & bmask], 1u);
  }
  __syncthreads();
  for (int b = threadIdx.x; b < 2048; b += 256) {
    const unsigned v = hs[b];
    if (v) atomicAdd(&ghist[row * 2048 + b], v);
  }
}

// ---------------- K5: pick the bin holding the k-th largest; zero bins ----------------
__global__ void k_scan(Sel* __restrict__ state, unsigned* __restrict__ ghist,
                       int shift, unsigned bmask, int nbins) {
  const int row = blockIdx.x;
  unsigned* h = ghist + row * 2048;
  if (threadIdx.x == 0) {
    Sel s = state[row];
    if (s.k > 0) {
      unsigned cum = 0, need = s.need, chosen = 0;
      bool found = false;
      for (int b = nbins - 1; b >= 0; --b) {
        const unsigned c = h[b];
        if (!found) {
          if (cum + c >= need) { chosen = (unsigned)b; s.need = need - cum; found = true; }
          cum += c;
        }
      }
      s.pval |= chosen << shift;
      s.pmask |= bmask << (unsigned)shift;
      state[row] = s;
    }
  }
  __syncthreads();
  for (int b = threadIdx.x; b < nbins; b += blockDim.x) h[b] = 0u;  // ready for next level
}

// ---------------- K6: count negatives above / equal to the threshold ----------------
__global__ void k_count(const float* __restrict__ ce, const Sel* __restrict__ state,
                        unsigned* __restrict__ cnt) {
  const int row = blockIdx.y;
  const int e0  = (blockIdx.x * 256 + threadIdx.x) * 8;
  const unsigned vg = state[row].pval;
  const float4* p = (const float4*)(ce + (size_t)row * LROW + e0);
  float4 v0 = p[0], v1 = p[1];
  const float f[8] = {v0.x, v0.y, v0.z, v0.w, v1.x, v1.y, v1.z, v1.w};
  unsigned gt = 0, eq = 0;
#pragma unroll
  for (int j = 0; j < 8; ++j) {
    const unsigned u = __float_as_uint(f[j]);
    const bool neg = (u >> 31) == 0u;        // target==0
    const unsigned bits = u & 0x7FFFFFFFu;
    if (neg) { gt += (bits > vg) ? 1u : 0u; eq += (bits == vg) ? 1u : 0u; }
  }
#pragma unroll
  for (int off = 16; off >= 1; off >>= 1) { gt += __shfl_xor(gt, off); eq += __shfl_xor(eq, off); }
  __shared__ unsigned rg[8], re[8];
  const int wave = threadIdx.x >> 5, lane = threadIdx.x & 31;
  if (lane == 0) { rg[wave] = gt; re[wave] = eq; }
  __syncthreads();
  if (threadIdx.x == 0) {
    unsigned a = 0, b = 0;
    for (int i = 0; i < 8; ++i) { a += rg[i]; b += re[i]; }
    atomicAdd(&cnt[row * 2], a);
    atomicAdd(&cnt[row * 2 + 1], b);
  }
}

// ---------------- K7: final scalar ----------------
__global__ void k_final(const float* __restrict__ part, const Sel* __restrict__ state,
                        const unsigned* __restrict__ cnt, float* __restrict__ out) {
  if (threadIdx.x == 0) {
    double s0 = 0.0, s1 = 0.0;
    for (int i = 0; i < 64; ++i) { s0 += part[i]; s1 += part[64 + i]; }
    s0 /= 16.0; s1 /= 16.0;                  // WMMA replicated each rowsum 16x
    double c = 0.0;
    for (int r = 0; r < NROWS; ++r) {
      const Sel s = state[r];
      if (s.k > 0) {
        const unsigned ngt = cnt[r * 2], neq = cnt[r * 2 + 1];
        const unsigned tie = s.need < neq ? s.need : neq;   // exact when threshold unique
        c += (double)(s.k + ngt + tie);      // |G U P| = k + (#neg in global top-k)
      }
    }
    const double NT = (double)NROWS * (double)LROW;
    out[0] = (float)((c * s1 + (NT - c) * s0) / ((double)LROW * NT));
  }
}

extern "C" void kernel_launch(void* const* d_in, const int* in_sizes, int n_in,
                              void* d_out, int out_size, void* d_ws, size_t ws_size,
                              hipStream_t stream) {
  const float*     logits = (const float*)d_in[0];
  const long long* tgt    = (const long long*)d_in[1];

  char* ws = (char*)d_ws;
  float*    ce   = (float*)ws;
  size_t off = (size_t)NROWS * LROW * sizeof(float);     // 75,497,472 B
  unsigned* pos  = (unsigned*)(ws + off); off += 256;
  Sel*      st   = (Sel*)(ws + off);      off += 1024;
  unsigned* hist = (unsigned*)(ws + off); off += (size_t)NROWS * 2048 * 4;
  unsigned* cnt  = (unsigned*)(ws + off); off += 512;
  float*    part = (float*)(ws + off);    off += 1024;

  k_zero<<<1, 256, 0, stream>>>(pos, hist, cnt, part);
  k_ce<<<dim3(NCHUNK, NROWS), 256, 0, stream>>>(logits, tgt, ce, pos);
  k_rowsum_wmma<<<dim3(48, 2), 256, 0, stream>>>(ce, part);
  k_sel_init<<<1, 32, 0, stream>>>(pos, st);
  // 3-level radix select over 31-bit ce bit patterns: 11 + 11 + 10 bits
  k_hist<<<dim3(NCHUNK, NROWS), 256, 0, stream>>>(ce, st, hist, 21, 0x7FFu);
  k_scan<<<NROWS, 64, 0, stream>>>(st, hist, 21, 0x7FFu, 2048);
  k_hist<<<dim3(NCHUNK, NROWS), 256, 0, stream>>>(ce, st, hist, 10, 0x7FFu);
  k_scan<<<NROWS, 64, 0, stream>>>(st, hist, 10, 0x7FFu, 2048);
  k_hist<<<dim3(NCHUNK, NROWS), 256, 0, stream>>>(ce, st, hist, 0, 0x3FFu);
  k_scan<<<NROWS, 64, 0, stream>>>(st, hist, 0, 0x3FFu, 1024);
  k_count<<<dim3(NCHUNK, NROWS), 256, 0, stream>>>(ce, st, cnt);
  k_final<<<1, 64, 0, stream>>>(part, st, cnt, (float*)d_out);
}
